// MGU_83210696392775
// MI455X (gfx1250) — matrix operand order repaired and checked by reading, not compiled, not run on hardware
//
#include <hip/hip_runtime.h>

// ---------------------------------------------------------------------------
// MGU RNN for MI455X (gfx1250, wave32, WMMA).
//  - All dense projections: bf16 A/B, f32 accumulate via v_wmma_f32_16x16x32_bf16.
//  - Sequential gate recurrence: one persistent 256-thread workgroup, h kept in
//    registers (wave-owned 32-column slices matching the WMMA C layout),
//    A-matrices (h and f*h, padded to 16 rows) staged in LDS.
// ---------------------------------------------------------------------------

typedef __attribute__((ext_vector_type(16))) __bf16 v16bf;
typedef __attribute__((ext_vector_type(8)))  __bf16 v8bf;
typedef __attribute__((ext_vector_type(8)))  float  v8f;

#define MGU_B 8
#define MGU_T 8192
#define MGU_H 256
#define MGU_L 2
#define MGU_M (MGU_B * MGU_T)   // 65536 rows

static __device__ __forceinline__ v8f vzero8() {
  v8f z = {0.f, 0.f, 0.f, 0.f, 0.f, 0.f, 0.f, 0.f};
  return z;
}

static __device__ __forceinline__ v16bf cat8(v8bf lo, v8bf hi) {
  return __builtin_shufflevector(lo, hi, 0, 1, 2, 3, 4, 5, 6, 7,
                                         8, 9, 10, 11, 12, 13, 14, 15);
}

// A fragment for v_wmma_f32_16x16x32_bf16 from a row-major [*, 256] bf16 buffer.
// Layout: lanes 0-15 hold row M=lane, K [0..15] contiguous;
// lanes 16-31 hold the same rows, K [16..31] (matches load_b_frag).
static __device__ __forceinline__ v16bf load_a_frag(const __bf16* __restrict__ base,
                                                    int row, int K0, int lh) {
  const __bf16* p = base + row * MGU_H + K0 + lh * 16;
  v8bf lo = *(const v8bf*)(p);
  v8bf hi = *(const v8bf*)(p + 8);
  return cat8(lo, hi);
}

// B fragment: B[k][n] = W[n][k] (Y = A * W^T). Lane holds column n0+(lane&15)
// and a contiguous 16-element K chunk starting at K0 + (lane>>4)*16.
static __device__ __forceinline__ v16bf load_b_frag(const __bf16* __restrict__ W,
                                                    int col, int K0, int lh) {
  return *(const v16bf*)(W + col * MGU_H + K0 + lh * 16);
}

// ---------------------------------------------------------------------------
// f32 -> bf16 conversion (n divisible by 4)
// ---------------------------------------------------------------------------
__global__ __launch_bounds__(256) void mgu_cvt_bf16(const float* __restrict__ s,
                                                    __bf16* __restrict__ d, int n) {
  int i = (blockIdx.x * blockDim.x + threadIdx.x) * 4;
  if (i < n) {
    float4 v = *(const float4*)(s + i);
    d[i + 0] = (__bf16)v.x;
    d[i + 1] = (__bf16)v.y;
    d[i + 2] = (__bf16)v.z;
    d[i + 3] = (__bf16)v.w;
  }
}

// ---------------------------------------------------------------------------
// GEMM: Y[M,256] = act(A[M,256](bf16) * W[256,256]^T(bf16) + bias)
// Block = 256 threads = 8 waves, covers 64 rows x 256 cols.
// Wave (w>>1) picks the 16-row M tile, (w&1) picks the 128-col half;
// each wave owns 8 N tiles of 16 (64 f32 accumulator VGPRs), K loop = 8x32.
// ---------------------------------------------------------------------------
template <int OUT_BF16, int LEAKY>
__global__ __launch_bounds__(256) void mgu_gemm(const __bf16* __restrict__ A,
                                                const __bf16* __restrict__ W,
                                                const float* __restrict__ bias,
                                                void* __restrict__ Yv) {
  const int wave = threadIdx.x >> 5;
  const int lane = threadIdx.x & 31;
  const int ln = lane & 15;
  const int lh = lane >> 4;
  const int mtile = blockIdx.x * 64 + (wave >> 1) * 16;
  const int nbase = (wave & 1) * 128;
  const int row = mtile + ln;

  v8f acc[8];
#pragma unroll
  for (int j = 0; j < 8; ++j) acc[j] = vzero8();

#pragma unroll
  for (int kk = 0; kk < 8; ++kk) {
    const int K0 = kk * 32;
    v16bf av = load_a_frag(A, row, K0, lh);
#pragma unroll
    for (int j = 0; j < 8; ++j) {
      v16bf bv = load_b_frag(W, nbase + j * 16 + ln, K0, lh);
      acc[j] = __builtin_amdgcn_wmma_f32_16x16x32_bf16(
          false, av, false, bv, (short)0, acc[j], false, false);
    }
  }

  float* Yf = (float*)Yv;
  __bf16* Yb = (__bf16*)Yv;
#pragma unroll
  for (int j = 0; j < 8; ++j) {
    const int col = nbase + j * 16 + ln;
    const float b = bias[col];
#pragma unroll
    for (int r = 0; r < 8; ++r) {
      const int m = mtile + r + lh * 8;   // C layout: row = r + 8*(lane>>4)
      float v = acc[j][r] + b;
      if (LEAKY) v = v > 0.f ? v : 0.01f * v;
      if (OUT_BF16) Yb[m * MGU_H + col] = (__bf16)v;
      else          Yf[m * MGU_H + col] = v;
    }
  }
}

// ---------------------------------------------------------------------------
// Persistent sequential MGU scan for one layer. grid = 1 block of 256 threads.
//   f     = sigmoid(xf_t + h @ Uf^T)
//   h_hat = tanh  (xh_t + (f*h) @ Uh^T)
//   h     = (1-f)*h + f*h_hat        (start-reset folded forward)
// Wave w owns output columns [32w, 32w+32): two 16x16 C tiles, 8+8 WMMAs per
// matvec (K=256). Batch rows 0-7 live in lanes 0-15 (pad rows 8-15 are zero).
// ---------------------------------------------------------------------------
__global__ __launch_bounds__(256) void mgu_scan(const __bf16* __restrict__ xf,
                                                const __bf16* __restrict__ xh,
                                                const unsigned char* __restrict__ start,
                                                const float* __restrict__ h0,
                                                const __bf16* __restrict__ Ufb,
                                                const __bf16* __restrict__ Uhb,
                                                __bf16* __restrict__ z,
                                                float* __restrict__ hfinal) {
  __shared__ __bf16 hA[16 * MGU_H];   // padded h (A of matmul 1)
  __shared__ __bf16 hB[16 * MGU_H];   // padded f*h (A of matmul 2)

  const int tid = threadIdx.x;
  for (int i = tid; i < 16 * MGU_H; i += 256) {
    hA[i] = (__bf16)0.f;
    hB[i] = (__bf16)0.f;
  }

  const int wave = tid >> 5;
  const int lane = tid & 31;
  const int ln = lane & 15;
  const int lh = lane >> 4;
  const int c0 = wave * 32 + ln;   // first owned column
  const int c1 = c0 + 16;          // second owned column

  float h0r[8], h1r[8], f0r[8], f1r[8];
#pragma unroll
  for (int r = 0; r < 8; ++r) { h0r[r] = h1r[r] = f0r[r] = f1r[r] = 0.f; }

  if (lh == 0) {
#pragma unroll
    for (int r = 0; r < 8; ++r) {
      float v0 = h0[r * MGU_H + c0];
      float v1 = h0[r * MGU_H + c1];
      if (start[r * MGU_T + 0]) { v0 = 0.f; v1 = 0.f; }
      h0r[r] = v0;
      h1r[r] = v1;
      hA[r * MGU_H + c0] = (__bf16)v0;
      hA[r * MGU_H + c1] = (__bf16)v1;
    }
  }
  __syncthreads();

  for (int t = 0; t < MGU_T; ++t) {
    // ---- phase 1: forget gate ------------------------------------------
    v8f a0 = vzero8(), a1 = vzero8();
#pragma unroll
    for (int kk = 0; kk < 8; ++kk) {
      const int K0 = kk * 32;
      v16bf av = load_a_frag(hA, ln, K0, lh);
      v16bf b0 = load_b_frag(Ufb, c0, K0, lh);
      v16bf b1 = load_b_frag(Ufb, c1, K0, lh);
      a0 = __builtin_amdgcn_wmma_f32_16x16x32_bf16(false, av, false, b0,
                                                   (short)0, a0, false, false);
      a1 = __builtin_amdgcn_wmma_f32_16x16x32_bf16(false, av, false, b1,
                                                   (short)0, a1, false, false);
    }
    if (lh == 0) {
#pragma unroll
      for (int r = 0; r < 8; ++r) {
        const int mrow = (r * MGU_T + t) * MGU_H;
        const float p0 = a0[r] + (float)xf[mrow + c0];
        const float p1 = a1[r] + (float)xf[mrow + c1];
        const float f0 = 1.f / (1.f + __expf(-p0));
        const float f1 = 1.f / (1.f + __expf(-p1));
        f0r[r] = f0;
        f1r[r] = f1;
        hB[r * MGU_H + c0] = (__bf16)(f0 * h0r[r]);
        hB[r * MGU_H + c1] = (__bf16)(f1 * h1r[r]);
      }
    }
    __syncthreads();

    // ---- phase 2: candidate + blend ------------------------------------
    a0 = vzero8();
    a1 = vzero8();
#pragma unroll
    for (int kk = 0; kk < 8; ++kk) {
      const int K0 = kk * 32;
      v16bf av = load_a_frag(hB, ln, K0, lh);
      v16bf b0 = load_b_frag(Uhb, c0, K0, lh);
      v16bf b1 = load_b_frag(Uhb, c1, K0, lh);
      a0 = __builtin_amdgcn_wmma_f32_16x16x32_bf16(false, av, false, b0,
                                                   (short)0, a0, false, false);
      a1 = __builtin_amdgcn_wmma_f32_16x16x32_bf16(false, av, false, b1,
                                                   (short)0, a1, false, false);
    }
    if (lh == 0) {
#pragma unroll
      for (int r = 0; r < 8; ++r) {
        const int mrow = (r * MGU_T + t) * MGU_H;
        const float hh0 = tanhf(a0[r] + (float)xh[mrow + c0]);
        const float hh1 = tanhf(a1[r] + (float)xh[mrow + c1]);
        float hn0 = (1.f - f0r[r]) * h0r[r] + f0r[r] * hh0;
        float hn1 = (1.f - f1r[r]) * h1r[r] + f1r[r] * hh1;
        z[mrow + c0] = (__bf16)hn0;          // z gets pre-reset value
        z[mrow + c1] = (__bf16)hn1;
        if (t + 1 < MGU_T && start[r * MGU_T + t + 1]) { hn0 = 0.f; hn1 = 0.f; }
        h0r[r] = hn0;
        h1r[r] = hn1;
        hA[r * MGU_H + c0] = (__bf16)hn0;
        hA[r * MGU_H + c1] = (__bf16)hn1;
      }
    }
    __syncthreads();
  }

  if (lh == 0) {
#pragma unroll
    for (int r = 0; r < 8; ++r) {
      hfinal[r * MGU_H + c0] = h0r[r];
      hfinal[r * MGU_H + c1] = h1r[r];
    }
  }
}

// ---------------------------------------------------------------------------
extern "C" void kernel_launch(void* const* d_in, const int* in_sizes, int n_in,
                              void* d_out, int out_size, void* d_ws, size_t ws_size,
                              hipStream_t stream) {
  const float* emb    = (const float*)d_in[0];
  const unsigned char* start = (const unsigned char*)d_in[1];  // jnp bool = 1 byte
  const float* h0     = (const float*)d_in[2];
  const float* min_w  = (const float*)d_in[3];
  const float* min_b  = (const float*)d_in[4];
  const float* mout_w = (const float*)d_in[5];
  const float* mout_b = (const float*)d_in[6];
  const float* Wf_w   = (const float*)d_in[7];
  const float* Wf_b   = (const float*)d_in[8];
  const float* Wh_w   = (const float*)d_in[9];
  const float* Wh_b   = (const float*)d_in[10];
  const float* Uf_w   = (const float*)d_in[11];
  const float* Uh_w   = (const float*)d_in[12];
  const float* ff_w   = (const float*)d_in[13];
  const float* ff_b   = (const float*)d_in[14];
  float* out = (float*)d_out;

  // Workspace layout (~130 MB): four 32MB bf16 activation buffers + weights.
  char* w = (char*)d_ws;
  const size_t MB32 = (size_t)32 << 20;
  __bf16* bufA = (__bf16*)(w);              // emb_bf, later z per layer
  __bf16* bufX = (__bf16*)(w + 1 * MB32);   // layer input x
  __bf16* bufF = (__bf16*)(w + 2 * MB32);   // xf
  __bf16* bufH = (__bf16*)(w + 3 * MB32);   // xh
  __bf16* wmin = (__bf16*)(w + 4 * MB32);
  __bf16* wmo  = wmin + 65536;
  __bf16* wWf  = wmo  + 65536;              // 2 layers
  __bf16* wWh  = wWf  + 131072;
  __bf16* wff  = wWh  + 131072;
  __bf16* wUf  = wff  + 131072;
  __bf16* wUh  = wUf  + 131072;

  const dim3 blk(256);

  // ---- precision conversions (weights stay L2-resident afterwards) -------
  mgu_cvt_bf16<<<dim3((MGU_M * MGU_H) / 4 / 256), blk, 0, stream>>>(emb, bufA, MGU_M * MGU_H);
  mgu_cvt_bf16<<<dim3(64),  blk, 0, stream>>>(min_w,  wmin, 65536);
  mgu_cvt_bf16<<<dim3(64),  blk, 0, stream>>>(mout_w, wmo,  65536);
  mgu_cvt_bf16<<<dim3(128), blk, 0, stream>>>(Wf_w,   wWf,  131072);
  mgu_cvt_bf16<<<dim3(128), blk, 0, stream>>>(Wh_w,   wWh,  131072);
  mgu_cvt_bf16<<<dim3(128), blk, 0, stream>>>(ff_w,   wff,  131072);
  mgu_cvt_bf16<<<dim3(128), blk, 0, stream>>>(Uf_w,   wUf,  131072);
  mgu_cvt_bf16<<<dim3(128), blk, 0, stream>>>(Uh_w,   wUh,  131072);

  const dim3 gg(MGU_M / 64);   // 1024 workgroups per GEMM

  // ---- map_in: x = emb @ min_w^T + min_b ----------------------------------
  mgu_gemm<1, 0><<<gg, blk, 0, stream>>>(bufA, wmin, min_b, bufX);

  for (int l = 0; l < MGU_L; ++l) {
    const int wo = l * 65536;   // per-layer 256x256 weight offset
    // time-parallel input projections
    mgu_gemm<1, 0><<<gg, blk, 0, stream>>>(bufX, wWf + wo, Wf_b + l * MGU_H, bufF);
    mgu_gemm<1, 0><<<gg, blk, 0, stream>>>(bufX, wWh + wo, Wh_b + l * MGU_H, bufH);
    // sequential recurrence (latency wall): 1 persistent workgroup
    mgu_scan<<<dim3(1), blk, 0, stream>>>(bufF, bufH, start,
                                          h0 + l * MGU_B * MGU_H,
                                          wUf + wo, wUh + wo,
                                          bufA,
                                          out + (size_t)MGU_M * MGU_H + l * MGU_B * MGU_H);
    // per-layer feed-forward + leaky_relu(0.01)
    mgu_gemm<1, 1><<<gg, blk, 0, stream>>>(bufA, wff + wo, ff_b + l * MGU_H, bufX);
  }

  // ---- map_out: f32 result straight into d_out ----------------------------
  mgu_gemm<0, 0><<<gg, blk, 0, stream>>>(bufX, wmo, mout_b, out);
}
